// Model_7971459301914
// MI455X (gfx1250) — matrix operand order repaired and checked by reading, not compiled, not run on hardware
//
#include <hip/hip_runtime.h>
#include <math.h>
#include <stdint.h>

#define D   768
#define BS  256
#define EPS 1e-05f
#define INV_T (1.0f / 0.03f)

typedef __bf16 bf16_t;
typedef __attribute__((ext_vector_type(16))) __bf16 v16bf;
typedef __attribute__((ext_vector_type(8)))  __bf16 v8bf;
typedef __attribute__((ext_vector_type(8)))  float  v8f;
typedef __attribute__((ext_vector_type(4)))  unsigned int u32x4;
typedef __attribute__((ext_vector_type(8)))  unsigned int u32x8;

// ---------------------------------------------------------------------------
// WMMA 16x16x32 bf16 lane layouts (ISA 7.12.2, wave32):
// A (16x32 MxK): lanes 0-15: M=lane, K = 0..15; lanes 16-31: M=lane-16,
//                K = 16..31 (contiguous, not the ISA table order).  -> two contiguous 8-element chunks.
// B (32x16 KxN, same K order as A): lanes 0-15: N=lane, K=0..15; lanes 16-31: N=lane-16, K=16..31.
//                -> one contiguous 16-element chunk (B stored transposed, NxK).
// C/D (16x16):   lanes 0-15: N=lane, M=vgpr; lanes 16-31: N=lane-16, M=vgpr+8.
// ---------------------------------------------------------------------------

// ---------------------------------------------------------------------------
// TDM: issue a 2D tensor_load_to_lds (bf16 tile rows x rowlen) per ISA ch.8.
// D# group0: count=1 | lds_addr | global_addr | type=2.
// D# group1: data_size=1(2B), tensor_dim0=rowlen, tensor_dim1=rows,
//            tile_dim0=rowlen, tile_dim1=rows, dim0_stride=stride.
// ---------------------------------------------------------------------------
__device__ __forceinline__ void tdm_load_2d_bf16(unsigned lds_off, const void* gptr,
                                                 unsigned rows, unsigned rowlen,
                                                 unsigned stride) {
  unsigned long long ga = (unsigned long long)(uintptr_t)gptr;
  u32x4 g0;
  g0[0] = 1u;                                            // count=1 (user D#)
  g0[1] = lds_off;                                       // lds_addr
  g0[2] = (unsigned)(ga & 0xFFFFFFFFu);                  // global_addr[31:0]
  g0[3] = (unsigned)((ga >> 32) & 0x01FFFFFFu) | (2u << 30); // addr[56:32] | type=2
  u32x8 g1;
  g1[0] = (1u << 16);                                    // data_size=1 -> 2 bytes
  g1[1] = (rowlen & 0xFFFFu) << 16;                      // tensor_dim0 lo16 @[63:48]
  g1[2] = ((rowlen >> 16) & 0xFFFFu) | ((rows & 0xFFFFu) << 16); // dim0 hi | dim1 lo
  g1[3] = ((rows >> 16) & 0xFFFFu) | ((rowlen & 0xFFFFu) << 16); // dim1 hi | tile_dim0
  g1[4] = rows & 0xFFFFu;                                // tile_dim1 (tile_dim2=0)
  g1[5] = stride;                                        // tensor_dim0_stride lo32
  g1[6] = 0u;
  g1[7] = 0u;
  asm volatile("tensor_load_to_lds %0, %1" :: "s"(g0), "s"(g1) : "memory");
}

// ---------------------------------------------------------------------------
// GEMM: C(f32, MxN) = A(bf16 MxK rm) * B^T, B stored as NxK row-major.
// 128 threads = 4 waves, one 16x16 C tile per wave. grid=(N/64, M/16).
// ---------------------------------------------------------------------------
__global__ __launch_bounds__(128)
void gemm_wmma_bf16t(const bf16_t* __restrict__ A, const bf16_t* __restrict__ B,
                     float* __restrict__ C, int M, int N, int K) {
  int lane = threadIdx.x & 31;
  int wave = threadIdx.x >> 5;
  int n0 = (blockIdx.x * 4 + wave) * 16;
  int m0 = blockIdx.y * 16;

  int n    = lane & 15;
  int kb_a = (lane < 16) ? 0 : 16;
  int kb_b = (lane < 16) ? 0 : 16;
  const bf16_t* pa = A + (size_t)(m0 + n) * K + kb_a;   // A row m = lane&15
  const bf16_t* pb = B + (size_t)(n0 + n) * K + kb_b;

  v8f acc = {};
  for (int k0 = 0; k0 < K; k0 += 32) {
    v8bf a_lo = *(const v8bf*)(pa + k0);        // K = kb_a + 0..7
    v8bf a_hi = *(const v8bf*)(pa + k0 + 8);    // K = kb_a + 8..15
    v16bf a;
#pragma unroll
    for (int e = 0; e < 8; ++e) { a[e] = a_lo[e]; a[8 + e] = a_hi[e]; }
    v16bf b = *(const v16bf*)(pb + k0);         // K = kb_b + 0..15 contiguous
    acc = __builtin_amdgcn_wmma_f32_16x16x32_bf16(false, a, false, b,
                                                  (short)0, acc, false, false);
  }
  int mb = (lane < 16) ? 0 : 8;
#pragma unroll
  for (int r = 0; r < 8; ++r)
    C[(size_t)(m0 + mb + r) * N + n0 + n] = acc[r];
}

// ---------------------------------------------------------------------------
// prep: row norms, bf16 conversions, pass-through outputs. grid=256, block=256
// ---------------------------------------------------------------------------
__global__ __launch_bounds__(256)
void prep_kernel(const float* __restrict__ code, const float* __restrict__ nl,
                 const float* __restrict__ aug,
                 bf16_t* __restrict__ code_bf, bf16_t* __restrict__ nl_bf,
                 bf16_t* __restrict__ cf_bf, bf16_t* __restrict__ nf_bf,
                 bf16_t* __restrict__ af_bf,
                 float* __restrict__ out_code, float* __restrict__ out_nl) {
  __shared__ float red0[256], red1[256], red2[256];
  int i = blockIdx.x, tid = threadIdx.x;
  float sc = 0.f, sn = 0.f, sa = 0.f;
  for (int c = tid; c < D; c += 256) {
    float vc = code[(size_t)i * D + c], vn = nl[(size_t)i * D + c], va = aug[(size_t)i * D + c];
    sc += vc * vc; sn += vn * vn; sa += va * va;
  }
  red0[tid] = sc; red1[tid] = sn; red2[tid] = sa;
  __syncthreads();
  for (int s = 128; s > 0; s >>= 1) {
    if (tid < s) {
      red0[tid] += red0[tid + s];
      red1[tid] += red1[tid + s];
      red2[tid] += red2[tid + s];
    }
    __syncthreads();
  }
  float rc = rsqrtf(red0[0]), rn = rsqrtf(red1[0]), ra = rsqrtf(red2[0]);
  for (int c = tid; c < D; c += 256) {
    size_t idx = (size_t)i * D + c;
    float vc = code[idx], vn = nl[idx], va = aug[idx];
    code_bf[idx] = (bf16_t)vc;
    nl_bf[idx]   = (bf16_t)vn;
    cf_bf[idx]   = (bf16_t)(vc * rc);
    nf_bf[idx]   = (bf16_t)(vn * rn);
    af_bf[idx]   = (bf16_t)(va * ra);
    out_code[idx] = vc;
    out_nl[idx]   = vn;
  }
}

// ---------------------------------------------------------------------------
// wprep: build TRANSPOSED bf16 weights (NxK) via LDS tile transpose:
//   WsumT[n][k]=Wa[k][n]+Wc[k][n], WdiffT[n][k]=Wb[k][n]-Wc[k][n], WdT[n][k]=Wd[k][n]
// grid=(24,24) of 32x32 tiles, block=256 (32x8)
// ---------------------------------------------------------------------------
__global__ __launch_bounds__(256)
void wprep_kernel(const float* __restrict__ W1, bf16_t* __restrict__ WsumT,
                  bf16_t* __restrict__ WdiffT, bf16_t* __restrict__ WdT) {
  __shared__ float ts[32][33], td[32][33], tw[32][33];
  int k0 = blockIdx.x * 32, c0 = blockIdx.y * 32;
  int tx = threadIdx.x & 31, ty = threadIdx.x >> 5;   // 32 x 8
  for (int rr = 0; rr < 32; rr += 8) {
    int k = k0 + ty + rr;
    float wa = W1[(size_t)k * D + c0 + tx];
    float wb = W1[(size_t)(D + k) * D + c0 + tx];
    float wc = W1[(size_t)(2 * D + k) * D + c0 + tx];
    float wd = W1[(size_t)(3 * D + k) * D + c0 + tx];
    ts[ty + rr][tx] = wa + wc;
    td[ty + rr][tx] = wb - wc;
    tw[ty + rr][tx] = wd;
  }
  __syncthreads();
  for (int rr = 0; rr < 32; rr += 8) {
    int n = c0 + ty + rr;
    WsumT [(size_t)n * D + k0 + tx] = (bf16_t)ts[tx][ty + rr];
    WdiffT[(size_t)n * D + k0 + tx] = (bf16_t)td[tx][ty + rr];
    WdT   [(size_t)n * D + k0 + tx] = (bf16_t)tw[tx][ty + rr];
  }
}

// ---------------------------------------------------------------------------
// pair_logits: dominant 77-GFLOP einsum. One WG per 16x16 (i,j) tile.
// nl/code tiles (16x768 bf16, 24KB each) staged into LDS by the TDM
// (tensor_load_to_lds + s_wait_tensorcnt), then 8 waves sweep 32-wide h slabs:
// per k-step one A fragment (E[jj,k]=nl[i,k]*code[jj,k], packed bf16 mul from
// LDS) feeds TWO WMMAs against contiguous WdT rows. tanh*W2 reduced with
// ds_add_f32 into a pair accumulator; sigmoid at the end.
// ---------------------------------------------------------------------------
__global__ __launch_bounds__(256)
void pair_logits_kernel(const bf16_t* __restrict__ nl_bf,
                        const bf16_t* __restrict__ code_bf,
                        const bf16_t* __restrict__ WdT,
                        const float* __restrict__ linA,
                        const float* __restrict__ linB,
                        const float* __restrict__ b1,
                        const float* __restrict__ W2,
                        const float* __restrict__ b2,
                        float* __restrict__ logits) {
  __shared__ __align__(16) bf16_t s_nl[16 * D];
  __shared__ __align__(16) bf16_t s_code[16 * D];
  __shared__ float s_acc[256];

  int i0 = blockIdx.y * 16;
  int j0 = blockIdx.x * 16;
  int tid = threadIdx.x;
  int lane = tid & 31;
  int wave = tid >> 5;

  if (wave == 0) {
    tdm_load_2d_bf16((unsigned)(uintptr_t)s_nl,   nl_bf   + (size_t)i0 * D, 16, D, D);
    tdm_load_2d_bf16((unsigned)(uintptr_t)s_code, code_bf + (size_t)j0 * D, 16, D, D);
    __builtin_amdgcn_s_wait_tensorcnt(0);
  }
  s_acc[tid] = 0.0f;
  __syncthreads();

  int n    = lane & 15;
  int kb_a = (lane < 16) ? 0 : 16;
  int kb_b = (lane < 16) ? 0 : 16;
  int mb   = (lane < 16) ? 0 : 8;

  const bf16_t* row_cd = s_code + (size_t)n * D + kb_a;   // A row m = jj = lane&15

  for (int pg = 0; pg < 16; ++pg) {                 // ii within tile
    const bf16_t* row_nl = s_nl + (size_t)pg * D + kb_a;
    for (int t = 0; t < 3; ++t) {                   // 3 slabs of 32 h per wave
      int h0 = (wave + 8 * t) * 32;
      const bf16_t* pb0 = WdT + (size_t)(h0 + n) * D + kb_b;
      const bf16_t* pb1 = WdT + (size_t)(h0 + 16 + n) * D + kb_b;
      v8f acc0 = {}, acc1 = {};
      for (int k0 = 0; k0 < D; k0 += 32) {
        v8bf p0 = *(const v8bf*)(row_nl + k0)      * *(const v8bf*)(row_cd + k0);
        v8bf p1 = *(const v8bf*)(row_nl + k0 + 8) * *(const v8bf*)(row_cd + k0 + 8);
        v16bf a;
#pragma unroll
        for (int e = 0; e < 8; ++e) { a[e] = p0[e]; a[8 + e] = p1[e]; }
        v16bf b0 = *(const v16bf*)(pb0 + k0);
        v16bf b1v = *(const v16bf*)(pb1 + k0);
        acc0 = __builtin_amdgcn_wmma_f32_16x16x32_bf16(false, a, false, b0,
                                                       (short)0, acc0, false, false);
        acc1 = __builtin_amdgcn_wmma_f32_16x16x32_bf16(false, a, false, b1v,
                                                       (short)0, acc1, false, false);
      }
      int i = i0 + pg;
#pragma unroll
      for (int half = 0; half < 2; ++half) {
        int h = h0 + half * 16 + n;
        float lA = linA[(size_t)i * D + h];
        float bb = b1[h];
        float w2 = W2[h];
#pragma unroll
        for (int r = 0; r < 8; ++r) {
          int jj = mb + r;
          float av = half ? acc1[r] : acc0[r];
          float v = av + lA + linB[(size_t)(j0 + jj) * D + h] + bb;
          atomicAdd(&s_acc[pg * 16 + jj], tanhf(v) * w2);   // ds_add_f32
        }
      }
    }
  }
  __syncthreads();

  int ii = tid >> 4, jj = tid & 15;
  float z = s_acc[tid] + b2[0];
  logits[(size_t)(i0 + ii) * BS + (j0 + jj)] = 1.0f / (1.0f + expf(-z));
}

// ---------------------------------------------------------------------------
// Block reductions (blockDim == 256)
// ---------------------------------------------------------------------------
__device__ __forceinline__ float blk_sum(float v, float* sh) {
  int tid = threadIdx.x;
  sh[tid] = v; __syncthreads();
  for (int s = 128; s > 0; s >>= 1) {
    if (tid < s) sh[tid] += sh[tid + s];
    __syncthreads();
  }
  float r = sh[0]; __syncthreads();
  return r;
}
__device__ __forceinline__ float blk_max(float v, float* sh) {
  int tid = threadIdx.x;
  sh[tid] = v; __syncthreads();
  for (int s = 128; s > 0; s >>= 1) {
    if (tid < s) sh[tid] = fmaxf(sh[tid], sh[tid + s]);
    __syncthreads();
  }
  float r = sh[0]; __syncthreads();
  return r;
}

// ---------------------------------------------------------------------------
// Per-row losses. grid=256 rows, block=256
// ---------------------------------------------------------------------------
__global__ __launch_bounds__(256)
void rowloss_kernel(const float* __restrict__ scores, const float* __restrict__ logits,
                    const float* __restrict__ cn, const float* __restrict__ cc,
                    const float* __restrict__ nn, const float* __restrict__ ca,
                    const float* __restrict__ na,
                    float* __restrict__ ce_row, float* __restrict__ code_row,
                    float* __restrict__ nl_row, float* __restrict__ neg_row,
                    float* __restrict__ pos_row) {
  __shared__ float sh[256];
  int i = blockIdx.x, j = threadIdx.x;

  // CE on scores
  float s  = scores[(size_t)i * BS + j];
  float mx = blk_max(s, sh);
  float se = blk_sum(expf(s - mx), sh);
  if (j == 0) ce_row[i] = scores[(size_t)i * BS + i] - (mx + logf(se));

  // CLS on logits
  float L  = logits[(size_t)i * BS + j];
  float ng = (j == i) ? 0.0f : logf(1.0f - L + EPS);
  float sn = blk_sum(ng, sh);
  if (j == 0) {
    neg_row[i] = sn;
    pos_row[i] = logf(logits[(size_t)i * BS + i] + EPS);
  }

  // CLR code side: [p | 0.8*n*(1-eye) | 0.1*pa]
  float v0 = cn[(size_t)i * BS + j] * INV_T;
  float v1 = (j == i) ? 0.0f : 0.8f * cc[(size_t)i * BS + j] * INV_T;
  float v2 = 0.1f * ca[(size_t)i * BS + j] * INV_T;
  float m3 = blk_max(fmaxf(v0, fmaxf(v1, v2)), sh);
  float s3 = blk_sum(expf(v0 - m3) + expf(v1 - m3) + expf(v2 - m3), sh);
  if (j == 0) {
    float d0 = cn[(size_t)i * BS + i] * INV_T;
    float d2 = 0.1f * ca[(size_t)i * BS + i] * INV_T;
    code_row[i] = logf(s3) - logf(expf(d0 - m3) + expf(d2 - m3));
  }

  // CLR nl side (nf@cf.T = cn transposed)
  float u0 = cn[(size_t)j * BS + i] * INV_T;
  float u1 = (j == i) ? 0.0f : 0.8f * nn[(size_t)i * BS + j] * INV_T;
  float u2 = 0.1f * na[(size_t)i * BS + j] * INV_T;
  float m4 = blk_max(fmaxf(u0, fmaxf(u1, u2)), sh);
  float s4 = blk_sum(expf(u0 - m4) + expf(u1 - m4) + expf(u2 - m4), sh);
  if (j == 0) {
    float d0 = cn[(size_t)i * BS + i] * INV_T;
    float d2 = 0.1f * na[(size_t)i * BS + i] * INV_T;
    nl_row[i] = logf(s4) - logf(expf(d0 - m4) + expf(d2 - m4));
  }
}

// ---------------------------------------------------------------------------
// Final combine -> d_out[0]
// ---------------------------------------------------------------------------
__global__ __launch_bounds__(256)
void final_kernel(const float* __restrict__ ce_row, const float* __restrict__ code_row,
                  const float* __restrict__ nl_row, const float* __restrict__ neg_row,
                  const float* __restrict__ pos_row, float* __restrict__ out) {
  __shared__ float sh[256];
  int tid = threadIdx.x;
  float s_ce = blk_sum(ce_row[tid], sh);
  float s_co = blk_sum(code_row[tid], sh);
  float s_nl = blk_sum(nl_row[tid], sh);
  float s_ng = blk_sum(neg_row[tid], sh);
  float s_ps = blk_sum(pos_row[tid], sh);
  if (tid == 0) {
    float loss_ce  = -(s_ce / (float)BS);
    float loss_clr = 0.5f * (s_co / (float)BS + s_nl / (float)BS);
    float loss_cls = -((s_ng / ((float)BS * (BS - 1))) + s_ps / (float)BS);
    out[0] = loss_ce + 0.1f * loss_clr + loss_cls;
  }
}

// ---------------------------------------------------------------------------
extern "C" void kernel_launch(void* const* d_in, const int* in_sizes, int n_in,
                              void* d_out, int out_size, void* d_ws, size_t ws_size,
                              hipStream_t stream) {
  (void)in_sizes; (void)n_in; (void)out_size; (void)ws_size;
  const float* code = (const float*)d_in[0];
  const float* nl   = (const float*)d_in[1];
  const float* aug  = (const float*)d_in[2];
  const float* W1   = (const float*)d_in[3];
  const float* b1   = (const float*)d_in[4];
  const float* W2   = (const float*)d_in[5];
  const float* b2   = (const float*)d_in[6];
  float* out = (float*)d_out;
  float* out_code = out + 1;
  float* out_nl   = out + 1 + (size_t)BS * D;

  char* w = (char*)d_ws;
  size_t off = 0;
  auto carve = [&](size_t bytes) -> char* {
    char* p = w + off;
    off = (off + bytes + 255) & ~(size_t)255;
    return p;
  };
  const size_t MATBF = (size_t)BS * D * 2;
  const size_t WBF   = (size_t)D * D * 2;
  const size_t MATF  = (size_t)BS * D * 4;
  const size_t SQF   = (size_t)BS * BS * 4;

  bf16_t* code_bf = (bf16_t*)carve(MATBF);
  bf16_t* nl_bf   = (bf16_t*)carve(MATBF);
  bf16_t* cf_bf   = (bf16_t*)carve(MATBF);
  bf16_t* nf_bf   = (bf16_t*)carve(MATBF);
  bf16_t* af_bf   = (bf16_t*)carve(MATBF);
  bf16_t* WsumT   = (bf16_t*)carve(WBF);
  bf16_t* WdiffT  = (bf16_t*)carve(WBF);
  bf16_t* WdT     = (bf16_t*)carve(WBF);
  float*  linA    = (float*)carve(MATF);
  float*  linB    = (float*)carve(MATF);
  float*  scores  = (float*)carve(SQF);
  float*  cn      = (float*)carve(SQF);
  float*  cc      = (float*)carve(SQF);
  float*  nn      = (float*)carve(SQF);
  float*  ca      = (float*)carve(SQF);
  float*  na      = (float*)carve(SQF);
  float*  logits  = (float*)carve(SQF);
  float*  ce_row   = (float*)carve(BS * 4);
  float*  code_row = (float*)carve(BS * 4);
  float*  nl_row   = (float*)carve(BS * 4);
  float*  neg_row  = (float*)carve(BS * 4);
  float*  pos_row  = (float*)carve(BS * 4);

  prep_kernel<<<BS, 256, 0, stream>>>(code, nl, aug, code_bf, nl_bf,
                                      cf_bf, nf_bf, af_bf, out_code, out_nl);
  wprep_kernel<<<dim3(D / 32, D / 32), 256, 0, stream>>>(W1, WsumT, WdiffT, WdT);

  // All GEMMs: A (MxK) * B^T with B stored NxK -> contiguous fragment loads.
  gemm_wmma_bf16t<<<dim3(D / 64, BS / 16), 128, 0, stream>>>(nl_bf,   WsumT,  linA,  BS, D,  D);
  gemm_wmma_bf16t<<<dim3(D / 64, BS / 16), 128, 0, stream>>>(code_bf, WdiffT, linB,  BS, D,  D);
  gemm_wmma_bf16t<<<dim3(BS / 64, BS / 16), 128, 0, stream>>>(nl_bf, code_bf, scores, BS, BS, D);
  gemm_wmma_bf16t<<<dim3(BS / 64, BS / 16), 128, 0, stream>>>(cf_bf, nf_bf,   cn,     BS, BS, D);
  gemm_wmma_bf16t<<<dim3(BS / 64, BS / 16), 128, 0, stream>>>(cf_bf, cf_bf,   cc,     BS, BS, D);
  gemm_wmma_bf16t<<<dim3(BS / 64, BS / 16), 128, 0, stream>>>(nf_bf, nf_bf,   nn,     BS, BS, D);
  gemm_wmma_bf16t<<<dim3(BS / 64, BS / 16), 128, 0, stream>>>(cf_bf, af_bf,   ca,     BS, BS, D);
  gemm_wmma_bf16t<<<dim3(BS / 64, BS / 16), 128, 0, stream>>>(nf_bf, af_bf,   na,     BS, BS, D);

  pair_logits_kernel<<<dim3(BS / 16, BS / 16), 256, 0, stream>>>(
      nl_bf, code_bf, WdT, linA, linB, b1, W2, b2, logits);

  rowloss_kernel<<<BS, 256, 0, stream>>>(scores, logits, cn, cc, nn, ca, na,
                                         ce_row, code_row, nl_row, neg_row, pos_row);
  final_kernel<<<1, 256, 0, stream>>>(ce_row, code_row, nl_row, neg_row, pos_row, out);
}
